// MyLLM_48000554500200
// MI455X (gfx1250) — matrix operand order repaired and checked by reading, not compiled, not run on hardware
//
#include <hip/hip_runtime.h>
#include <hip/hip_bf16.h>
#include <math.h>

// ---------------- model dims (fixed by reference) ----------------
#define B_    2
#define T_    1024
#define D_    1024
#define H_    16
#define HD_   64
#define L_    8
#define DFF_  4096
#define V_    32000
#define NROW  (B_*T_)      // 2048
#define QKVW  (3*D_)       // 3072

// ---------------- WMMA types ----------------
typedef __attribute__((ext_vector_type(16))) __bf16 v16bf;
typedef __attribute__((ext_vector_type(8)))  float  v8f;

union Frag {
    v16bf v;
    uint4 q[2];
    unsigned short s[16];
};

__device__ __forceinline__ v8f vzero8() {
    v8f z = {0.f,0.f,0.f,0.f,0.f,0.f,0.f,0.f};
    return z;
}

// fp32 -> bf16 via native convert (RNE); gfx1250 has hardware bf16 cvt
__device__ __forceinline__ unsigned short f2bf(float f) {
    __bf16 h = (__bf16)f;
    return __builtin_bit_cast(unsigned short, h);
}
__device__ __forceinline__ float bf2f(unsigned short s) {
    union { unsigned int u; float f; } v; v.u = ((unsigned int)s) << 16;
    return v.f;
}

__device__ __forceinline__ v8f wmma_bf16(const Frag& A, const Frag& B, v8f C) {
    // D = A(16x32 bf16) * B(32x16 bf16) + C(16x16 f32)
    return __builtin_amdgcn_wmma_f32_16x16x32_bf16(
        false, A.v, false, B.v, (short)0, C, false, false);
}

// ---------------- embedding + positional encoding ----------------
__global__ __launch_bounds__(256) void embed_k(const int* __restrict__ idx,
                                               const float* __restrict__ emb,
                                               float* __restrict__ x) {
    int gid = blockIdx.x * 256 + threadIdx.x;       // over NROW*D_
    int row = gid >> 10;            // D_=1024
    int d   = gid & 1023;
    int t   = row & (T_ - 1);
    int tok = idx[row];
    int i2  = d & ~1;
    float div = __expf((float)i2 * (-9.210340371976184f / (float)D_)); // ln(10000)
    float ang = (float)t * div;
    float pe  = (d & 1) ? __cosf(ang) : __sinf(ang);
    x[(size_t)row * D_ + d] = emb[(size_t)tok * D_ + d] + pe;
}

// ---------------- LayerNorm (one block per row) ----------------
__global__ __launch_bounds__(256) void layernorm_k(const float* __restrict__ x,
                                                   const float* __restrict__ g,
                                                   const float* __restrict__ bb,
                                                   float* __restrict__ y) {
    __shared__ float rs_[8], rq_[8], stat[2];
    int row = blockIdx.x, tid = threadIdx.x;
    const float4* xr = (const float4*)(x + (size_t)row * D_);
    float4 v = xr[tid];                               // 256 threads * 4 = 1024
    float s = v.x + v.y + v.z + v.w;
    float q = v.x*v.x + v.y*v.y + v.z*v.z + v.w*v.w;
    #pragma unroll
    for (int o = 16; o > 0; o >>= 1) {
        s += __shfl_down(s, o, 32);
        q += __shfl_down(q, o, 32);
    }
    if ((tid & 31) == 0) { rs_[tid >> 5] = s; rq_[tid >> 5] = q; }
    __syncthreads();
    if (tid == 0) {
        float S = 0.f, Q = 0.f;
        #pragma unroll
        for (int i = 0; i < 8; ++i) { S += rs_[i]; Q += rq_[i]; }
        float mu  = S * (1.0f / D_);
        float var = Q * (1.0f / D_) - mu * mu;
        stat[0] = mu;
        stat[1] = rsqrtf(var + 1e-5f);
    }
    __syncthreads();
    float mu = stat[0], r = stat[1];
    float4 gg = ((const float4*)g)[tid];
    float4 bv = ((const float4*)bb)[tid];
    float4 o;
    o.x = (v.x - mu) * r * gg.x + bv.x;
    o.y = (v.y - mu) * r * gg.y + bv.y;
    o.z = (v.z - mu) * r * gg.z + bv.z;
    o.w = (v.w - mu) * r * gg.w + bv.w;
    ((float4*)(y + (size_t)row * D_))[tid] = o;
}

// ---------------- bf16-WMMA GEMM: C = A(MxK) * B(KxN) [+bias][+res][ReLU] ----
// Block: 256 thr = 8 waves, tile 64(M) x 128(N), K-step 32.
// Wave (wid/2 -> M strip of 16, wid&1 -> N strip of 64; 4 WMMA tiles/wave).
template<bool BIAS, bool RELU, bool RES>
__global__ __launch_bounds__(256) void gemm_bf16(const float* __restrict__ A,
                                                 const float* __restrict__ Bm,
                                                 const float* __restrict__ bias,
                                                 const float* __restrict__ res,
                                                 float* __restrict__ Cm,
                                                 int M, int N, int K) {
    __shared__ unsigned short As[64][40];    // padded: stride 80B -> 16B aligned, 20-bank step
    __shared__ unsigned short Bs[128][40];   // stored transposed: Bs[n][k]
    const int tid  = threadIdx.x;
    const int wid  = tid >> 5, lane = tid & 31;
    const int lh   = lane >> 4, lm = lane & 15;
    const int bm   = blockIdx.y * 64, bn = blockIdx.x * 128;
    const int wm   = (wid >> 1) * 16, wn = (wid & 1) * 64;

    v8f acc[4];
    #pragma unroll
    for (int i = 0; i < 4; ++i) acc[i] = vzero8();

    const int nk = K >> 5;
    for (int kb = 0; kb < nk; ++kb) {
        const int k0 = kb << 5;
        __syncthreads();
        // stage A: 64x32 fp32 -> bf16 (coalesced rows of 32)
        #pragma unroll
        for (int it = 0; it < 8; ++it) {
            int i2 = tid + it * 256;
            int r = i2 >> 5, c = i2 & 31;
            As[r][c] = f2bf(A[(size_t)(bm + r) * K + k0 + c]);
        }
        // stage B: 32x128 fp32 (coalesced) -> Bs[n][k] bf16 (20-bank step, conflict-free)
        #pragma unroll
        for (int it = 0; it < 16; ++it) {
            int i2 = tid + it * 256;
            int k = i2 >> 7, n = i2 & 127;
            Bs[n][k] = f2bf(Bm[(size_t)(k0 + k) * N + bn + n]);
        }
        // prefetch next K-step's weight panel (global_prefetch_b8 path)
        if (kb + 1 < nk) {
            int i2 = tid;                    // one float per thread: only rows k0+32, k0+33 of the panel
            int k = i2 >> 7, n = i2 & 127;
            __builtin_prefetch(&Bm[(size_t)(k0 + 32 + k) * N + bn + n], 0, 1);
        }
        __syncthreads();

        // A fragment: lane (lh,lm): row wm+lm, K = {8lh+0..7, 16+8lh+0..7}
        Frag a;
        a.q[0] = *(const uint4*)&As[wm + lm][8 * lh];
        a.q[1] = *(const uint4*)&As[wm + lm][16 + 8 * lh];
        #pragma unroll
        for (int nt = 0; nt < 4; ++nt) {
            // B fragment: col wn+nt*16+lm, K = {8lh+0..7, 16+8lh+0..7}
            Frag b;
            const unsigned short* bp = &Bs[wn + nt * 16 + lm][8 * lh];
            b.q[0] = ((const uint4*)bp)[0];
            b.q[1] = ((const uint4*)bp)[2];
            acc[nt] = wmma_bf16(a, b, acc[nt]);
        }
    }

    // epilogue: C layout per lane: M = i + 8*lh, N = lm
    #pragma unroll
    for (int nt = 0; nt < 4; ++nt) {
        int col = bn + wn + nt * 16 + lm;
        float bv = BIAS ? bias[col] : 0.0f;
        #pragma unroll
        for (int i = 0; i < 8; ++i) {
            int row = bm + wm + i + 8 * lh;
            float v = acc[nt][i] + bv;
            if (RELU) v = v > 0.f ? v : 0.f;
            if (RES)  v += res[(size_t)row * N + col];
            Cm[(size_t)row * N + col] = v;
        }
    }
}

// ---------------- fused causal attention ----------------
// One block (128 thr = 4 waves) per (b, h, 16-query tile).
// S[16][1032] bf16 scores (pad 8: A-frag reads conflict-free, 16B aligned).
__global__ __launch_bounds__(128) void attn_k(const float* __restrict__ qkv,
                                              float* __restrict__ out) {
    __shared__ unsigned short S[16][1032];
    __shared__ unsigned short qs[16][72];
    const int qt  = blockIdx.x, h = blockIdx.y, bz = blockIdx.z;
    const int tid = threadIdx.x, wid = tid >> 5, lane = tid & 31;
    const int lh  = lane >> 4, lm = lane & 15;
    const int qbase = qt * 16;
    const int ktA   = (qt + 2) & ~1;           // needed key tiles, rounded to even
    const float scale = 0.125f;                // 1/sqrt(64)

    const float* qptr = qkv + (size_t)(bz * T_) * QKVW + h * HD_;
    const float* kptr = qptr + D_;
    const float* vptr = qptr + 2 * D_;

    // stage q tile 16x64 -> bf16
    #pragma unroll
    for (int it = 0; it < 8; ++it) {
        int i2 = tid + it * 128;
        int r = i2 >> 6, c = i2 & 63;
        qs[r][c] = f2bf(qptr[(size_t)(qbase + r) * QKVW + c]);
    }
    __syncthreads();

    // q fragments (reused across all key tiles): HD=64 -> two 16x32 slabs
    Frag a0, a1;
    a0.q[0] = *(const uint4*)&qs[lm][8 * lh];
    a0.q[1] = *(const uint4*)&qs[lm][16 + 8 * lh];
    a1.q[0] = *(const uint4*)&qs[lm][32 + 8 * lh];
    a1.q[1] = *(const uint4*)&qs[lm][48 + 8 * lh];

    // ---- phase 1: scores (each wave owns key tiles wid, wid+4, ...) ----
    for (int kt = wid; kt < ktA; kt += 4) {
        int kbase = kt * 16;
        // B frag: key = kbase+lm, dims {8lh+0..7, 16+8lh+0..7} (b0) / +32 (b1)
        const float* kv = kptr + (size_t)(kbase + lm) * QKVW;
        Frag b0, b1;
        #pragma unroll
        for (int q4 = 0; q4 < 4; ++q4) {
            float4 f0 = ((const float4*)kv)[2 * lh + q4 + (q4 >> 1) * 2];
            float4 f1 = ((const float4*)kv)[8 + 2 * lh + q4 + (q4 >> 1) * 2];
            b0.s[4*q4+0] = f2bf(f0.x); b0.s[4*q4+1] = f2bf(f0.y);
            b0.s[4*q4+2] = f2bf(f0.z); b0.s[4*q4+3] = f2bf(f0.w);
            b1.s[4*q4+0] = f2bf(f1.x); b1.s[4*q4+1] = f2bf(f1.y);
            b1.s[4*q4+2] = f2bf(f1.z); b1.s[4*q4+3] = f2bf(f1.w);
        }
        v8f c = vzero8();
        c = wmma_bf16(a0, b0, c);
        c = wmma_bf16(a1, b1, c);
        #pragma unroll
        for (int i = 0; i < 8; ++i) {
            int m = i + 8 * lh;                 // query row in tile
            int kcol = kbase + lm;              // key index
            float sv = (kcol <= qbase + m) ? c[i] * scale : -INFINITY;
            S[m][kcol] = f2bf(sv);
        }
    }
    __syncthreads();

    // ---- phase 2: softmax (8 lanes per row, shuffle butterfly) ----
    {
        int m = tid >> 3, sub = tid & 7;
        int ncol = ktA * 16;
        float mx = -INFINITY;
        for (int c0 = sub; c0 < ncol; c0 += 8) mx = fmaxf(mx, bf2f(S[m][c0]));
        #pragma unroll
        for (int o = 1; o < 8; o <<= 1) mx = fmaxf(mx, __shfl_xor(mx, o, 32));
        float sum = 0.f;
        for (int c0 = sub; c0 < ncol; c0 += 8) sum += __expf(bf2f(S[m][c0]) - mx);
        #pragma unroll
        for (int o = 1; o < 8; o <<= 1) sum += __shfl_xor(sum, o, 32);
        float inv = 1.0f / sum;
        for (int c0 = sub; c0 < ncol; c0 += 8)
            S[m][c0] = f2bf(__expf(bf2f(S[m][c0]) - mx) * inv);
    }
    __syncthreads();

    // ---- phase 3: O = P * V; wave wid owns HD columns [16*wid, 16*wid+16) ----
    {
        int nb = wid * 16;
        v8f c = vzero8();
        int nch = ktA >> 1;
        for (int ch = 0; ch < nch; ++ch) {
            int kk = ch * 32;
            Frag a, b;
            a.q[0] = *(const uint4*)&S[lm][kk + 8 * lh];
            a.q[1] = *(const uint4*)&S[lm][kk + 16 + 8 * lh];
            #pragma unroll
            for (int j = 0; j < 16; ++j)
                b.s[j] = f2bf(vptr[(size_t)(kk + 8 * lh + j + (j >> 3) * 8) * QKVW + nb + lm]);
            c = wmma_bf16(a, b, c);
        }
        #pragma unroll
        for (int i = 0; i < 8; ++i) {
            int m = i + 8 * lh;
            out[(size_t)(bz * T_ + qbase + m) * D_ + h * HD_ + nb + lm] = c[i];
        }
    }
}

// ---------------- host-side orchestration ----------------
extern "C" void kernel_launch(void* const* d_in, const int* in_sizes, int n_in,
                              void* d_out, int out_size, void* d_ws, size_t ws_size,
                              hipStream_t stream) {
    (void)in_sizes; (void)n_in; (void)out_size; (void)ws_size;
    const int*   idx   = (const int*)  d_in[0];
    const float* emb   = (const float*)d_in[1];
    const float* Wqkv  = (const float*)d_in[2];
    const float* Wproj = (const float*)d_in[3];
    const float* bproj = (const float*)d_in[4];
    const float* ln1g  = (const float*)d_in[5];
    const float* ln1b  = (const float*)d_in[6];
    const float* ln2g  = (const float*)d_in[7];
    const float* ln2b  = (const float*)d_in[8];
    const float* W1    = (const float*)d_in[9];
    const float* b1    = (const float*)d_in[10];
    const float* W2    = (const float*)d_in[11];
    const float* b2    = (const float*)d_in[12];
    const float* lnfg  = (const float*)d_in[13];
    const float* lnfb  = (const float*)d_in[14];
    const float* Whead = (const float*)d_in[15];

    float* ws  = (float*)d_ws;
    float* x   = ws;                         //  2M f32
    float* xn  = x   + (size_t)NROW * D_;    //  2M
    float* qkv = xn  + (size_t)NROW * D_;    //  6M
    float* ao  = qkv + (size_t)NROW * QKVW;  //  2M
    float* ffh = ao  + (size_t)NROW * D_;    //  8M   (total 80 MB)

    embed_k<<<(NROW * D_) / 256, 256, 0, stream>>>(idx, emb, x);

    for (int l = 0; l < L_; ++l) {
        layernorm_k<<<NROW, 256, 0, stream>>>(x, ln1g + l * D_, ln1b + l * D_, xn);
        gemm_bf16<false, false, false>
            <<<dim3(QKVW / 128, NROW / 64), 256, 0, stream>>>(
                xn, Wqkv + (size_t)l * D_ * QKVW, nullptr, nullptr,
                qkv, NROW, QKVW, D_);
        attn_k<<<dim3(T_ / 16, H_, B_), 128, 0, stream>>>(qkv, ao);
        gemm_bf16<true, false, true>
            <<<dim3(D_ / 128, NROW / 64), 256, 0, stream>>>(
                ao, Wproj + (size_t)l * D_ * D_, bproj + (size_t)l * D_,
                x, x, NROW, D_, D_);
        layernorm_k<<<NROW, 256, 0, stream>>>(x, ln2g + l * D_, ln2b + l * D_, xn);
        gemm_bf16<true, true, false>
            <<<dim3(DFF_ / 128, NROW / 64), 256, 0, stream>>>(
                xn, W1 + (size_t)l * D_ * DFF_, b1 + (size_t)l * DFF_,
                nullptr, ffh, NROW, DFF_, D_);
        gemm_bf16<true, false, true>
            <<<dim3(D_ / 128, NROW / 64), 256, 0, stream>>>(
                ffh, W2 + (size_t)l * DFF_ * D_, b2 + (size_t)l * D_,
                x, x, NROW, D_, DFF_);
    }

    layernorm_k<<<NROW, 256, 0, stream>>>(x, lnfg, lnfb, xn);
    gemm_bf16<false, false, false>
        <<<dim3(V_ / 128, NROW / 64), 256, 0, stream>>>(
            xn, Whead, nullptr, nullptr, (float*)d_out, NROW, V_, D_);
}
